// KBCModel_1537598292519
// MI455X (gfx1250) — matrix operand order repaired and checked by reading, not compiled, not run on hardware
//
#include <hip/hip_runtime.h>

// Problem constants (match reference).
constexpr int BDIM = 2048;
constexpr int DDIM = 512;
constexpr int NDIM = 100000;
constexpr int FDIM = 64;

// GEMM tiling.
constexpr int BM = 256;    // rows per block (16 waves x 16)
constexpr int BN = 32;     // cols per block (2 x 16 per wave)
constexpr int KC = 128;    // K staged in LDS per chunk
constexpr int KP = KC + 8; // padded row (272B stride -> conflict-free b128 reads)
constexpr int NC = DDIM / KC; // 4 chunks

typedef __attribute__((ext_vector_type(16))) __bf16 v16bf;
typedef __attribute__((ext_vector_type(8)))  float  v8f;
typedef __attribute__((ext_vector_type(4)))  unsigned int u32x4;
typedef __attribute__((ext_vector_type(8)))  int i32x8;
typedef __attribute__((ext_vector_type(4)))  int i32x4;

union BFrag {
    v16bf v;
    uint4 q[2];
};

__device__ __forceinline__ unsigned short f2bf_rne(float f) {
    unsigned u = __builtin_bit_cast(unsigned, f);
    unsigned r = u + 0x7FFFu + ((u >> 16) & 1u);
    return (unsigned short)(r >> 16);
}
__device__ __forceinline__ float bf2f(unsigned short h) {
    unsigned u = ((unsigned)h) << 16;
    return __builtin_bit_cast(float, u);
}

// 3-term split-bf16 fp32 emulation micro-kernel: hi*hi + hi*lo + lo*hi
// for two N-subtiles sharing the A fragments.
__device__ __forceinline__ void wmma_6(const unsigned short* pah,
                                       const unsigned short* pal,
                                       const unsigned short* bh0,
                                       const unsigned short* bh1,
                                       const unsigned short* bl0,
                                       const unsigned short* bl1,
                                       v8f& acc0, v8f& acc1) {
    BFrag a_hi, a_lo, b0h, b1h, b0l, b1l;
    a_hi.q[0] = *(const uint4*)(pah);
    a_hi.q[1] = *(const uint4*)(pah + 8);
    a_lo.q[0] = *(const uint4*)(pal);
    a_lo.q[1] = *(const uint4*)(pal + 8);
    b0h.q[0] = *(const uint4*)(bh0);  b0h.q[1] = *(const uint4*)(bh0 + 8);
    b1h.q[0] = *(const uint4*)(bh1);  b1h.q[1] = *(const uint4*)(bh1 + 8);
    b0l.q[0] = *(const uint4*)(bl0);  b0l.q[1] = *(const uint4*)(bl0 + 8);
    b1l.q[0] = *(const uint4*)(bl1);  b1l.q[1] = *(const uint4*)(bl1 + 8);
    acc0 = __builtin_amdgcn_wmma_f32_16x16x32_bf16(false, a_hi.v, false, b0h.v, (short)0, acc0, false, false);
    acc1 = __builtin_amdgcn_wmma_f32_16x16x32_bf16(false, a_hi.v, false, b1h.v, (short)0, acc1, false, false);
    acc0 = __builtin_amdgcn_wmma_f32_16x16x32_bf16(false, a_hi.v, false, b0l.v, (short)0, acc0, false, false);
    acc1 = __builtin_amdgcn_wmma_f32_16x16x32_bf16(false, a_hi.v, false, b1l.v, (short)0, acc1, false, false);
    acc0 = __builtin_amdgcn_wmma_f32_16x16x32_bf16(false, a_lo.v, false, b0h.v, (short)0, acc0, false, false);
    acc1 = __builtin_amdgcn_wmma_f32_16x16x32_bf16(false, a_lo.v, false, b1h.v, (short)0, acc1, false, false);
}

// C/D layout: lane = N col (ll), VGPR v = row M = v + 8*lh.
__device__ __forceinline__ void count_epilogue(const v8f& acc0, const v8f& acc1,
                                               const float* __restrict__ targets,
                                               int* __restrict__ counts,
                                               int wm, int lh, int ll) {
    int cnt[8];
#pragma unroll
    for (int v = 0; v < 8; ++v) {
        const float tgt = targets[wm + v + lh * 8];
        cnt[v] = (acc0[v] >= tgt ? 1 : 0) + (acc1[v] >= tgt ? 1 : 0);
    }
#pragma unroll
    for (int o = 1; o < 16; o <<= 1) {
#pragma unroll
        for (int v = 0; v < 8; ++v) cnt[v] += __shfl_xor(cnt[v], o, 32);
    }
    if (ll == 0) {
#pragma unroll
        for (int v = 0; v < 8; ++v)
            atomicAdd(&counts[wm + v + lh * 8], cnt[v]);
    }
}

// TDM: one DMA pulls a [BN rows][KC halves] bf16 tile (row stride DDIM halves)
// into LDS with 16B padding every 256B row -> KP=136 padded layout.
__device__ __forceinline__ void tdm_load_tile_bf16(unsigned lds_off,
                                                   const unsigned short* gsrc) {
    const unsigned long long ga = (unsigned long long)(const void*)gsrc;
    u32x4 g0 = {0u, 0u, 0u, 0u};
    g0[0] = 1u;                                 // count=1 (valid user descriptor)
    g0[1] = lds_off;                            // lds_addr
    g0[2] = (unsigned)ga;                       // global_addr[31:0]
    g0[3] = (unsigned)(ga >> 32) | (2u << 30);  // global_addr[56:32] | type=2
    i32x8 g1 = {0, 0, 0, 0, 0, 0, 0, 0};
    g1[0] = (1 << 16)    // data_size = 2 bytes
          | (1 << 20)    // pad_enable
          | (5 << 22)    // pad_interval: 64 DWORDs (one 256B row)
          | (3 << 25);   // pad_amount: 4 DWORDs (16B) -> 272B row stride
    // tensor_dim0/1 large (in-bounds tiles only); dim0 bits79:48, dim1 bits111:80
    g1[2] = (int)((1u << 20) >> 16);                               // dim0[31:16]
    g1[3] = (int)(((1u << 20) >> 16) | ((unsigned)KC << 16));      // dim1[31:16] | tile_dim0
    g1[4] = BN;                                                    // tile_dim1 | tile_dim2=0
    g1[5] = DDIM;                                                  // tensor_dim0_stride
    const i32x4 z4 = {0, 0, 0, 0};
#if defined(__clang_major__) && __clang_major__ >= 23
    const i32x8 z8 = {0, 0, 0, 0, 0, 0, 0, 0};
    __builtin_amdgcn_tensor_load_to_lds(g0, g1, z4, z4, z8, 0);
#else
    __builtin_amdgcn_tensor_load_to_lds(g0, g1, z4, z4, 0);
#endif
}

// ---------------------------------------------------------------------------
// Kernel 1: per-row prep. Exact fp32 target, bf16 hi/lo split of q,
// and mask corrections pre-subtracted from the count (with dedup).
// ---------------------------------------------------------------------------
__global__ __launch_bounds__(128) void kbc_prep_kernel(
    const float* __restrict__ q, const float* __restrict__ rhs,
    const int* __restrict__ queries, const int* __restrict__ filt,
    float* __restrict__ targets, int* __restrict__ counts,
    unsigned short* __restrict__ qhi, unsigned short* __restrict__ qlo) {
    const int b = blockIdx.x;
    const int t = threadIdx.x;

    __shared__ float red[128];
    __shared__ int   cand[FDIM + 1];
    __shared__ int   validv[FDIM + 1];
    __shared__ float starget;
    __shared__ int   ccnt;

    const int tru = queries[b * 3 + 2];

    float part = 0.f;
    for (int d = t; d < DDIM; d += 128) {
        const float x = q[(size_t)b * DDIM + d];
        const unsigned short h = f2bf_rne(x);
        qhi[(size_t)b * DDIM + d] = h;
        qlo[(size_t)b * DDIM + d] = f2bf_rne(x - bf2f(h));
        part += x * rhs[(size_t)d * NDIM + tru];
    }
    red[t] = part;
    if (t == 0) ccnt = 0;
    if (t < FDIM) cand[t] = filt[b * FDIM + t];
    if (t == FDIM) cand[FDIM] = tru;
    __syncthreads();
    for (int s = 64; s > 0; s >>= 1) {
        if (t < s) red[t] += red[t + s];
        __syncthreads();
    }
    if (t == 0) { starget = red[0]; targets[b] = red[0]; }
    if (t <= FDIM) {
        int v = 1;
        for (int c = 0; c < t; ++c)
            if (cand[c] == cand[t]) { v = 0; break; }
        validv[t] = v;
    }
    __syncthreads();
    int local = 0;
    if (t <= FDIM && validv[t]) {
        const int j = cand[t];
        if (j == tru) {
            local = 1;
        } else {
            float s = 0.f;
            for (int d = 0; d < DDIM; ++d)
                s += q[(size_t)b * DDIM + d] * rhs[(size_t)d * NDIM + j];
            local = (s >= starget) ? 1 : 0;
        }
    }
    if (local) atomicAdd(&ccnt, local);
    __syncthreads();
    if (t == 0) counts[b] = -ccnt;
}

// ---------------------------------------------------------------------------
// Kernel 2a (fast path): transpose rhs and split to bf16 hi/lo once:
// rhsT[n][k]. Read coalesced along n, transpose through LDS, packed b32 writes.
// ---------------------------------------------------------------------------
__global__ __launch_bounds__(256) void kbc_split_transpose_kernel(
    const float* __restrict__ rhs,
    unsigned short* __restrict__ hiT,
    unsigned short* __restrict__ loT) {
    __shared__ unsigned short sh[32][34];
    __shared__ unsigned short sl[32][34];
    const int n0 = blockIdx.x * 32;
    const int k0 = blockIdx.y * 32;
    const int tid = threadIdx.x;
    const int tj = tid & 31;   // n-local
    const int ti = tid >> 5;   // k-local base
#pragma unroll
    for (int p = 0; p < 4; ++p) {
        const int k = ti + p * 8;
        const float x = rhs[(size_t)(k0 + k) * NDIM + (n0 + tj)];
        const unsigned short h = f2bf_rne(x);
        sh[tj][k] = h;
        sl[tj][k] = f2bf_rne(x - bf2f(h));
    }
    __syncthreads();
#pragma unroll
    for (int p = 0; p < 2; ++p) {
        const int idx = p * 256 + tid;  // 0..511
        const int r   = idx >> 4;       // n-local row
        const int c2  = idx & 15;       // u32 column
        const unsigned hv = (unsigned)sh[r][2 * c2] | ((unsigned)sh[r][2 * c2 + 1] << 16);
        const unsigned lv = (unsigned)sl[r][2 * c2] | ((unsigned)sl[r][2 * c2 + 1] << 16);
        ((unsigned*)(hiT + (size_t)(n0 + r) * DDIM + k0))[c2] = hv;
        ((unsigned*)(loT + (size_t)(n0 + r) * DDIM + k0))[c2] = lv;
    }
}

// ---------------------------------------------------------------------------
// Kernel 2b (fast path): WMMA GEMM + rank counting, B tiles staged by the
// Tensor Data Mover (double-buffered, overlapped via TENSORcnt).
// ---------------------------------------------------------------------------
__global__ __launch_bounds__(512) void kbc_wmma_rank_tdm_kernel(
    const unsigned short* __restrict__ rhsT_hi,
    const unsigned short* __restrict__ rhsT_lo,
    const unsigned short* __restrict__ qhi,
    const unsigned short* __restrict__ qlo,
    const float* __restrict__ targets,
    int* __restrict__ counts) {
    __shared__ __align__(16) unsigned short bh_s[2][BN][KP];
    __shared__ __align__(16) unsigned short bl_s[2][BN][KP];

    const int n0   = blockIdx.x * BN;
    const int m0   = blockIdx.y * BM;
    const int tid  = threadIdx.x;
    const int wave = tid >> 5;
    const int lane = tid & 31;
    const int lh   = lane >> 4;
    const int ll   = lane & 15;
    const int wm   = m0 + wave * 16;

    const size_t bbase = (size_t)n0 * DDIM;

    if (wave == 0) {
        tdm_load_tile_bf16((unsigned)(size_t)&bh_s[0][0][0], rhsT_hi + bbase);
        tdm_load_tile_bf16((unsigned)(size_t)&bl_s[0][0][0], rhsT_lo + bbase);
        __builtin_amdgcn_s_wait_tensorcnt(0);
    }
    __syncthreads();

    v8f acc0 = {};
    v8f acc1 = {};

#pragma unroll
    for (int c = 0; c < NC; ++c) {
        const int cur = c & 1;
        const int nxt = cur ^ 1;
        if (wave == 0 && c + 1 < NC) {
            tdm_load_tile_bf16((unsigned)(size_t)&bh_s[nxt][0][0],
                               rhsT_hi + bbase + (size_t)(c + 1) * KC);
            tdm_load_tile_bf16((unsigned)(size_t)&bl_s[nxt][0][0],
                               rhsT_lo + bbase + (size_t)(c + 1) * KC);
        }
        const int k0 = c * KC;
#pragma unroll
        for (int ks = 0; ks < KC; ks += 32) {
            const int kk = k0 + ks + lh * 16;
            const int kb = ks + lh * 16;
            wmma_6(qhi + (size_t)(wm + ll) * DDIM + kk,
                   qlo + (size_t)(wm + ll) * DDIM + kk,
                   &bh_s[cur][ll][kb], &bh_s[cur][16 + ll][kb],
                   &bl_s[cur][ll][kb], &bl_s[cur][16 + ll][kb],
                   acc0, acc1);
        }
        if (wave == 0 && c + 1 < NC) __builtin_amdgcn_s_wait_tensorcnt(0);
        __syncthreads();
    }

    count_epilogue(acc0, acc1, targets, counts, wm, lh, ll);
}

// ---------------------------------------------------------------------------
// Kernel 2c (fallback, small workspace): in-kernel fp32->bf16 split staging.
// ---------------------------------------------------------------------------
__global__ __launch_bounds__(512) void kbc_wmma_rank_conv_kernel(
    const float* __restrict__ rhs,
    const unsigned short* __restrict__ qhi,
    const unsigned short* __restrict__ qlo,
    const float* __restrict__ targets,
    int* __restrict__ counts) {
    __shared__ __align__(16) unsigned short bh_s[BN][KP];
    __shared__ __align__(16) unsigned short bl_s[BN][KP];

    const int n0   = blockIdx.x * BN;
    const int m0   = blockIdx.y * BM;
    const int tid  = threadIdx.x;
    const int wave = tid >> 5;
    const int lane = tid & 31;
    const int lh   = lane >> 4;
    const int ll   = lane & 15;
    const int wm   = m0 + wave * 16;

    v8f acc0 = {};
    v8f acc1 = {};

    for (int k0 = 0; k0 < DDIM; k0 += KC) {
        __syncthreads();
        // Stage rhs chunk: each thread converts a k-pair, one packed b32 store.
#pragma unroll
        for (int p = 0; p < (KC * BN) / (2 * 512); ++p) {
            const int e  = p * 512 + tid;
            const int n  = e & (BN - 1);
            const int k  = (e >> 5) * 2;
            const float x0 = rhs[(size_t)(k0 + k) * NDIM + (n0 + n)];
            const float x1 = rhs[(size_t)(k0 + k + 1) * NDIM + (n0 + n)];
            const unsigned short h0 = f2bf_rne(x0);
            const unsigned short h1 = f2bf_rne(x1);
            const unsigned short l0 = f2bf_rne(x0 - bf2f(h0));
            const unsigned short l1 = f2bf_rne(x1 - bf2f(h1));
            *(unsigned*)&bh_s[n][k] = (unsigned)h0 | ((unsigned)h1 << 16);
            *(unsigned*)&bl_s[n][k] = (unsigned)l0 | ((unsigned)l1 << 16);
        }
        __syncthreads();

#pragma unroll
        for (int ks = 0; ks < KC; ks += 32) {
            const int kk = k0 + ks + lh * 16;
            const int kb = ks + lh * 16;
            wmma_6(qhi + (size_t)(wm + ll) * DDIM + kk,
                   qlo + (size_t)(wm + ll) * DDIM + kk,
                   &bh_s[ll][kb], &bh_s[16 + ll][kb],
                   &bl_s[ll][kb], &bl_s[16 + ll][kb],
                   acc0, acc1);
        }
    }

    count_epilogue(acc0, acc1, targets, counts, wm, lh, ll);
}

// ---------------------------------------------------------------------------
// Kernel 3: rank[b] = 1 + count[b]
// ---------------------------------------------------------------------------
__global__ void kbc_finalize_kernel(const int* __restrict__ counts,
                                    float* __restrict__ out) {
    const int b = blockIdx.x * blockDim.x + threadIdx.x;
    if (b < BDIM) out[b] = 1.0f + (float)counts[b];
}

extern "C" void kernel_launch(void* const* d_in, const int* in_sizes, int n_in,
                              void* d_out, int out_size, void* d_ws, size_t ws_size,
                              hipStream_t stream) {
    const float* q       = (const float*)d_in[0];
    const float* rhs     = (const float*)d_in[1];
    const int*   queries = (const int*)d_in[2];
    const int*   filt    = (const int*)d_in[3];
    float*       out     = (float*)d_out;

    // Workspace layout:
    //   targets (8KB) | counts (8KB) | q_hi (2MB) | q_lo (2MB)
    //   [fast path] rhsT_hi (102.4MB) | rhsT_lo (102.4MB)
    char* ws = (char*)d_ws;
    float* targets = (float*)ws;
    int*   counts  = (int*)(ws + BDIM * sizeof(float));
    unsigned short* qhi = (unsigned short*)(ws + 2 * BDIM * sizeof(float));
    unsigned short* qlo = qhi + (size_t)BDIM * DDIM;
    unsigned short* rhsT_hi = qlo + (size_t)BDIM * DDIM;
    unsigned short* rhsT_lo = rhsT_hi + (size_t)NDIM * DDIM;

    const size_t ws_base = 2 * BDIM * sizeof(float) + 2 * (size_t)BDIM * DDIM * 2;
    const size_t ws_full = ws_base + 2 * (size_t)NDIM * DDIM * 2;

    kbc_prep_kernel<<<BDIM, 128, 0, stream>>>(q, rhs, queries, filt,
                                              targets, counts, qhi, qlo);
    if (ws_size >= ws_full) {
        kbc_split_transpose_kernel<<<dim3(NDIM / 32, DDIM / 32), 256, 0, stream>>>(
            rhs, rhsT_hi, rhsT_lo);
        kbc_wmma_rank_tdm_kernel<<<dim3(NDIM / BN, BDIM / BM), 512, 0, stream>>>(
            rhsT_hi, rhsT_lo, qhi, qlo, targets, counts);
    } else {
        kbc_wmma_rank_conv_kernel<<<dim3(NDIM / BN, BDIM / BM), 512, 0, stream>>>(
            rhs, qhi, qlo, targets, counts);
    }
    kbc_finalize_kernel<<<(BDIM + 255) / 256, 256, 0, stream>>>(counts, out);
}
